// DTCSTGCN_48541720379761
// MI455X (gfx1250) — matrix operand order repaired and checked by reading, not compiled, not run on hardware
//
#include <hip/hip_runtime.h>
#include <hip/hip_bf16.h>

// Problem constants
#define B_ 8
#define T_ 24
#define N_ 512
#define LSTMH 64

typedef __attribute__((ext_vector_type(16))) __bf16 v16bf;
typedef __attribute__((ext_vector_type(8)))  float  v8f;

// ---------------- WMMA helpers (gfx1250 wave32 layouts) ----------------

__device__ inline v8f wmma_bf16(v16bf a, v16bf b, v8f c) {
#if defined(__gfx1250__)
  return __builtin_amdgcn_wmma_f32_16x16x32_bf16(false, a, false, b, (short)0, c, false, false);
#else
  (void)a; (void)b; return c;   // host-pass placeholder, never executed
#endif
}

// A (16x32 bf16, M x K) from LDS, row-major with leading dim lda.
// lane<16: row=lane, K = {0..15}; lane>=16: row=lane-16, K = {16..31}
__device__ inline v16bf load_a_lds(const __bf16* sA, int lda, int kOff, int lane) {
  int row = lane & 15, hi = lane >> 4;
  const __bf16* p = sA + row * lda + kOff + hi * 16;
  v16bf a;
#pragma unroll
  for (int e = 0; e < 8; ++e) { a[e] = p[e]; a[e + 8] = p[e + 8]; }
  return a;
}

// Packed B operand: for a K x C matrix, fragment (mc, ct) holds the 32x16 tile
// K = mc*32 + (lane>=16)*16 + e, C = ct*16 + (lane&15), stored so each lane's
// 16 bf16 elements are CONTIGUOUS (32 B -> 2x b128, coalesced across the wave).
// flat index = (((ct*kTiles + mc)*32) + lane)*16 + e
__device__ inline v16bf load_b_packed(const __bf16* Bp, int kTiles, int mc, int ct, int lane) {
  const __bf16* p = Bp + (((size_t)ct * kTiles + mc) * 32 + lane) * 16;
  v16bf b;
#pragma unroll
  for (int e = 0; e < 16; ++e) b[e] = p[e];
  return b;
}

// ---------------- small prep kernels ----------------

// u[n] = sum_m Wa[m,n] * wswp[m],  wswp[k] = sum_n Ws[n]*Wp[n,k]
__global__ void prep_u_kernel(const float* Ws, const float* Wp, const float* Wa, float* u) {
  __shared__ float wswp[N_];
  int k = threadIdx.x;                       // 512 threads
  float s = 0.f;
  for (int n = 0; n < N_; ++n) s += Ws[n] * Wp[n * N_ + k];
  wswp[k] = s;
  __syncthreads();
  float su = 0.f;
  for (int m = 0; m < N_; ++m) su += Wa[m * N_ + k] * wswp[m];
  u[k] = su;
}

// Vs (512x512 f32, row-major K x C) -> packed WMMA-B bf16 layout
__global__ void pack_vs_kernel(const float* Vs, __bf16* Vsp) {
  int idx = blockIdx.x * 256 + threadIdx.x;  // < 512*512
  int e    = idx & 15;
  int lane = (idx >> 4) & 31;
  int tile = idx >> 9;                       // ct*16 + mc  (ct<32, mc<16)
  int mc = tile & 15, ct = tile >> 4;
  int k = mc * 32 + (lane >> 4) * 16 + e;
  int c = ct * 16 + (lane & 15);
  Vsp[idx] = (__bf16)Vs[k * N_ + c];
}

// G1[b,t][k,c] = x[b,t,k]*W1[c] + b1[c], written directly in packed-B layout
__global__ void build_g1_kernel(const float* x, const float* W1, const float* b1, __bf16* G1) {
  int idx = blockIdx.x * 256 + threadIdx.x;  // < B*T*N*32
  int e    = idx & 15;
  int lane = (idx >> 4) & 31;
  int tile = (idx >> 9) & 31;                // per-bt: ct*16+mc (ct<2, mc<16)
  int bt   = idx >> 14;                      // 512*32 = 16384 per (b,t)
  int mc = tile & 15, ct = tile >> 4;
  int k = mc * 32 + (lane >> 4) * 16 + e;
  int c = ct * 16 + (lane & 15);
  G1[idx] = (__bf16)(x[(size_t)bt * N_ + k] * W1[c] + b1[c]);
}

// dis[b,t,i] = rsqrt(sum_j mask[i,j]*exp(-(f_i-f_j)^2))
__global__ void dis_kernel(const float* x, const float* mask, float* dis) {
  __shared__ float sf[N_];
  int bt = blockIdx.x;                       // b*24+t
  const float* f = x + (size_t)bt * N_;
  for (int j = threadIdx.x; j < N_; j += 256) sf[j] = f[j];
  __syncthreads();
  for (int i = threadIdx.x; i < N_; i += 256) {
    float fi = sf[i];
    const float* mrow = mask + (size_t)i * N_;
    float deg = 0.f;
    for (int j = 0; j < N_; ++j) {
      float m = mrow[j];
      if (m != 0.f) { float d = fi - sf[j]; deg += m * __expf(-d * d); }
    }
    dis[(size_t)bt * N_ + i] = deg > 0.f ? rsqrtf(deg) : 0.f;
  }
}

// v[b,t,k] = dis_k * sum_n (u[n]*dis_n)*mask[k,n]*exp(-(f_k-f_n)^2), t=0..22
__global__ void v_kernel(const float* x, const float* mask, const float* dis,
                         const float* u, float* v) {
  __shared__ float sf[N_], sw[N_];
  int bidx = blockIdx.x;                     // b*23+t
  int b = bidx / 23, t = bidx % 23;
  int bt = b * T_ + t;
  const float* f = x + (size_t)bt * N_;
  const float* d = dis + (size_t)bt * N_;
  for (int j = threadIdx.x; j < N_; j += 256) { sf[j] = f[j]; sw[j] = u[j] * d[j]; }
  __syncthreads();
  for (int k = threadIdx.x; k < N_; k += 256) {
    float fk = sf[k];
    const float* mrow = mask + (size_t)k * N_;  // mask symmetric
    float s = 0.f;
    for (int n = 0; n < N_; ++n) {
      float m = mrow[n];
      if (m != 0.f) { float dd = fk - sf[n]; s += m * sw[n] * __expf(-dd * dd); }
    }
    v[(size_t)bidx * N_ + k] = d[k] * s;
  }
}

// ---------------- fused Pt + H1 kernel (the 49 GFLOP GEMM) ----------------
// One block = (b, t in 0..22, 16-row tile). Computes:
//   S = sigmoid(fp.v + bs) @ Vs   (WMMA, K=512)  -> softmax rows -> Pt
//   H1[b,t+1,tile,:32] = lrelu((At[b,t+1]*Pt) @ G1[b,t+1])   (WMMA, K=512)
__global__ void __launch_bounds__(256) pt_h1_kernel(
    const float* x, const float* mask, const float* dis, const float* v,
    const float* bs, const __bf16* Vsp, const __bf16* G1, float* H1)
{
  __shared__ __bf16 sA[16 * N_];   // sigmoid(pre) tile, later reused as At*Pt
  __shared__ float  sS[16 * N_];   // S tile, later exp values
  __shared__ float  sv[N_], sf1[N_], sd1[N_];
  __shared__ float  red[256];
  __shared__ float  rowmax[16], rowinv[16], sfp[16];

  int it   = blockIdx.x & 31;
  int bt23 = blockIdx.x >> 5;      // b*23+t
  int b = bt23 / 23, t = bt23 % 23;
  int i0 = it * 16;
  int tid = threadIdx.x, lane = tid & 31, wave = tid >> 5;

  const float* fT   = x   + ((size_t)b * T_ + t) * N_;
  const float* fT1  = x   + ((size_t)b * T_ + t + 1) * N_;
  const float* dT1  = dis + ((size_t)b * T_ + t + 1) * N_;
  const float* vrow = v   + (size_t)bt23 * N_;

  for (int j = tid; j < N_; j += 256) { sv[j] = vrow[j]; sf1[j] = fT1[j]; sd1[j] = dT1[j]; }
  if (tid < 16) sfp[tid] = fT[i0 + tid];
  __syncthreads();

  // phase 1: sigmoid(pre) tile -> bf16 LDS
  for (int e = tid; e < 16 * N_; e += 256) {
    int r = e >> 9, m = e & 511;
    float pre = sfp[r] * sv[m] + bs[(size_t)(i0 + r) * N_ + m];
    sA[e] = (__bf16)(1.f / (1.f + __expf(-pre)));
  }
  __syncthreads();

  // phase 2: S = sigP @ Vs.  mc-outer / kt-inner: each A fragment loaded once,
  // 4 independent WMMAs issue back-to-back per wave.
  {
    v8f acc[4] = {};
    for (int mc = 0; mc < 16; ++mc) {
      v16bf a = load_a_lds(sA, N_, mc * 32, lane);
#pragma unroll
      for (int q = 0; q < 4; ++q) {
        v16bf bb = load_b_packed(Vsp, 16, mc, wave + q * 8, lane);
        acc[q] = wmma_bf16(a, bb, acc[q]);
      }
    }
    int col = lane & 15, rhi = (lane >> 4) * 8;
#pragma unroll
    for (int q = 0; q < 4; ++q) {
      int kt = wave + q * 8;
#pragma unroll
      for (int g = 0; g < 8; ++g) sS[(g + rhi) * N_ + kt * 16 + col] = acc[q][g];
    }
  }
  __syncthreads();

  // phase 3: row softmax over k (16 threads per row)
  int r = tid >> 4, sub = tid & 15;
  {
    float mx = -3.4e38f;
    const float* rowp = sS + r * N_ + sub * 32;
    for (int q = 0; q < 32; ++q) mx = fmaxf(mx, rowp[q]);
    red[tid] = mx;
  }
  __syncthreads();
  if (sub == 0) {
    float mx = red[r * 16];
    for (int q = 1; q < 16; ++q) mx = fmaxf(mx, red[r * 16 + q]);
    rowmax[r] = mx;
  }
  __syncthreads();
  {
    float mx = rowmax[r], s = 0.f;
    float* rowp = sS + r * N_ + sub * 32;
    for (int q = 0; q < 32; ++q) { float e2 = __expf(rowp[q] - mx); rowp[q] = e2; s += e2; }
    red[tid] = s;
  }
  __syncthreads();
  if (sub == 0) {
    float s = 0.f;
    for (int q = 0; q < 16; ++q) s += red[r * 16 + q];
    rowinv[r] = 1.f / s;
  }
  __syncthreads();

  // phase 4: A_att = At[b,t+1] * Pt  -> bf16 (recompute At on the fly)
  for (int e = tid; e < 16 * N_; e += 256) {
    int rr = e >> 9, k = e & 511;
    int i = i0 + rr;
    float m = mask[(size_t)i * N_ + k];
    float a = 0.f;
    if (m != 0.f) { float dd = sf1[i] - sf1[k]; a = m * __expf(-dd * dd) * sd1[i] * sd1[k]; }
    sA[e] = (__bf16)(a * sS[e] * rowinv[rr]);
  }
  __syncthreads();

  // phase 5: H1 tile = lrelu(A_att @ G1[b,t+1])  (2 waves, 16-col tiles)
  if (wave < 2) {
    const __bf16* Gp = G1 + ((size_t)b * T_ + t + 1) * N_ * 32;
    v8f acc = {};
    for (int mc = 0; mc < 16; ++mc) {
      v16bf a  = load_a_lds(sA, N_, mc * 32, lane);
      v16bf bb = load_b_packed(Gp, 16, mc, wave, lane);
      acc = wmma_bf16(a, bb, acc);
    }
    int col = lane & 15, rhi = (lane >> 4) * 8;
#pragma unroll
    for (int g = 0; g < 8; ++g) {
      float y = acc[g];
      y = y < 0.f ? 0.01f * y : y;
      H1[(((size_t)b * T_ + t + 1) * N_ + i0 + g + rhi) * 32 + wave * 16 + col] = y;
    }
  }
}

// ---------------- generic GCN propagation: out = lrelu(At @ G), C=32/64 ----------------
template <int C>
__global__ void __launch_bounds__(128) gcn_prop_kernel(
    const float* x, const float* mask, const float* dis,
    const __bf16* G, float* out, int tStart, int tCount)
{
  __shared__ __bf16 sA[16 * N_];
  __shared__ float sf[N_], sd[N_];
  int it  = blockIdx.x & 31;
  int lin = blockIdx.x >> 5;
  int b = lin / tCount, t = tStart + (lin % tCount);
  int i0 = it * 16;
  int tid = threadIdx.x, lane = tid & 31, wave = tid >> 5;
  size_t bt = (size_t)b * T_ + t;
  const float* f = x + bt * N_;
  const float* d = dis + bt * N_;
  for (int j = tid; j < N_; j += 128) { sf[j] = f[j]; sd[j] = d[j]; }
  __syncthreads();
  for (int e = tid; e < 16 * N_; e += 128) {
    int r = e >> 9, k = e & 511;
    int i = i0 + r;
    float m = mask[(size_t)i * N_ + k];
    float a = 0.f;
    if (m != 0.f) { float dd = sf[i] - sf[k]; a = m * __expf(-dd * dd) * sd[i] * sd[k]; }
    sA[e] = (__bf16)a;
  }
  __syncthreads();
  if (wave < C / 16) {
    const __bf16* Gp = G + bt * N_ * C;
    v8f acc = {};
    for (int mc = 0; mc < 16; ++mc) {
      v16bf a  = load_a_lds(sA, N_, mc * 32, lane);
      v16bf bb = load_b_packed(Gp, 16, mc, wave, lane);
      acc = wmma_bf16(a, bb, acc);
    }
    int col = lane & 15, rhi = (lane >> 4) * 8;
#pragma unroll
    for (int g = 0; g < 8; ++g) {
      float y = acc[g];
      y = y < 0.f ? 0.01f * y : y;
      out[(bt * N_ + i0 + g + rhi) * C + wave * 16 + col] = y;
    }
  }
}

// G2[bt][k,c] = H1[btk,:]@W2[:,c] + b2[c], written in packed-B layout
__global__ void build_g2_kernel(const float* H1, const float* W2, const float* b2, __bf16* G2) {
  __shared__ float sW[32 * 32];
  __shared__ float sb[32];
  int tid = threadIdx.x;
  for (int j = tid; j < 1024; j += 256) sW[j] = W2[j];
  if (tid < 32) sb[tid] = b2[tid];
  __syncthreads();
  int row = blockIdx.x * 256 + tid;         // btk, < B*T*N
  int bt = row >> 9, k = row & 511;
  int mc = k >> 5, hi = (k >> 4) & 1, e = k & 15;
  float h[32];
  const float* hp = H1 + (size_t)row * 32;
#pragma unroll
  for (int q = 0; q < 32; ++q) h[q] = hp[q];
  __bf16* gbase = G2 + (size_t)bt * (N_ * 32);
  for (int c = 0; c < 32; ++c) {
    float s = sb[c];
#pragma unroll
    for (int q = 0; q < 32; ++q) s += h[q] * sW[q * 32 + c];
    int ct = c >> 4, col = c & 15;
    gbase[(((ct * 16 + mc) * 32) + hi * 16 + col) * 16 + e] = (__bf16)s;
  }
}

// G3[bt][k,c] = H2[btk,:]@W3[:,c] + b3[c], c<64, packed-B layout
__global__ void build_g3_kernel(const float* H2, const float* W3, const float* b3, __bf16* G3) {
  __shared__ float sW[32 * 64];
  __shared__ float sb[64];
  int tid = threadIdx.x;
  for (int j = tid; j < 2048; j += 256) sW[j] = W3[j];
  if (tid < 64) sb[tid] = b3[tid];
  __syncthreads();
  int row = blockIdx.x * 256 + tid;
  int bt = row >> 9, k = row & 511;
  int mc = k >> 5, hi = (k >> 4) & 1, e = k & 15;
  float h[32];
  const float* hp = H2 + (size_t)row * 32;
#pragma unroll
  for (int q = 0; q < 32; ++q) h[q] = hp[q];
  __bf16* gbase = G3 + (size_t)bt * (N_ * 64);
  for (int c = 0; c < 64; ++c) {
    float s = sb[c];
#pragma unroll
    for (int q = 0; q < 32; ++q) s += h[q] * sW[q * 64 + c];
    int ct = c >> 4, col = c & 15;
    gbase[(((ct * 16 + mc) * 32) + hi * 16 + col) * 16 + e] = (__bf16)s;
  }
}

__global__ void mean_h3_kernel(const float* H3, float* Hs) {
  int idx = blockIdx.x * 256 + threadIdx.x;  // < B*N*64
  int c = idx & 63;
  int n = (idx >> 6) & 511;
  int b = idx >> 15;
  float s = 0.f;
  for (int t = 0; t < T_; ++t) s += H3[(((size_t)b * T_ + t) * N_ + n) * 64 + c];
  Hs[idx] = s * (1.f / 24.f);
}

// ---------------- LSTM: one block (64 threads) per (b,n) row ----------------
__global__ void __launch_bounds__(64) lstm_kernel(
    const float* x, const float* W_ih, const float* W_hh,
    const float* b_ih, const float* b_hh, float* Ft)
{
  __shared__ float sh[LSTMH];
  int row = blockIdx.x;                      // b*512+n
  int b = row >> 9, n = row & 511;
  int tid = threadIdx.x;
  float c = 0.f;
  sh[tid] = 0.f;
  float wih[4], bsum[4];
#pragma unroll
  for (int q = 0; q < 4; ++q) {
    int gi = q * LSTMH + tid;
    wih[q]  = W_ih[gi];
    bsum[q] = b_ih[gi] + b_hh[gi];
  }
  __syncthreads();
  for (int t = 0; t < T_; ++t) {
    float xv = x[((size_t)b * T_ + t) * N_ + n];
    float g4[4];
#pragma unroll
    for (int q = 0; q < 4; ++q) {
      int gi = q * LSTMH + tid;
      const float* wrow = W_hh + (size_t)gi * LSTMH;
      float s = xv * wih[q] + bsum[q];
      for (int k = 0; k < LSTMH; ++k) s += wrow[k] * sh[k];
      g4[q] = s;
    }
    float ig = 1.f / (1.f + __expf(-g4[0]));
    float fg = 1.f / (1.f + __expf(-g4[1]));
    float gg = tanhf(g4[2]);
    float og = 1.f / (1.f + __expf(-g4[3]));
    c = fg * c + ig * gg;
    float hn = og * tanhf(c);
    __syncthreads();
    sh[tid] = hn;
    __syncthreads();
  }
  Ft[(size_t)row * LSTMH + tid] = sh[tid];
}

// ---------------- attention + gating: Fa = Fo * softmax(tanh(Fo@Wa+b)) ----------------
__global__ void __launch_bounds__(128) attn_kernel(
    const float* Hs, const float* Ft, const float* W_attn, const float* b_attn, float* Fa)
{
  __shared__ float sFo[128];
  __shared__ float red[128];
  int row = blockIdx.x;                      // b*512+n
  int c = threadIdx.x;
  float fo = (c < 64) ? Hs[(size_t)row * 64 + c] : Ft[(size_t)row * 64 + (c - 64)];
  sFo[c] = fo;
  __syncthreads();
  float z = b_attn[c];
  for (int k = 0; k < 128; ++k) z += sFo[k] * W_attn[k * 128 + c];
  z = tanhf(z);
  red[c] = z;
  __syncthreads();
  for (int off = 64; off > 0; off >>= 1) {
    if (c < off) red[c] = fmaxf(red[c], red[c + off]);
    __syncthreads();
  }
  float mx = red[0];
  __syncthreads();
  float e = __expf(z - mx);
  red[c] = e;
  __syncthreads();
  for (int off = 64; off > 0; off >>= 1) {
    if (c < off) red[c] += red[c + off];
    __syncthreads();
  }
  Fa[(size_t)row * 128 + c] = fo * (e / red[0]);
}

__global__ void gh1_kernel(const float* Fa, const float* Wh1, const float* bh1, float* Gh1) {
  int idx = blockIdx.x * 256 + threadIdx.x;  // < B*N*32
  int c = idx & 31, bn = idx >> 5;
  const float* fp = Fa + (size_t)bn * 128;
  float s = bh1[c];
  for (int k = 0; k < 128; ++k) s += fp[k] * Wh1[k * 32 + c];
  Gh1[idx] = s;
}

template <int C>
__global__ void __launch_bounds__(64) hprop_kernel(const float* adj, const float* G, float* out) {
  __shared__ float arow[N_];
  int b = blockIdx.x >> 9, i = blockIdx.x & 511;
  int tid = threadIdx.x;
  const float* ap = adj + (size_t)i * N_;
  for (int j = tid; j < N_; j += 64) arow[j] = ap[j];
  __syncthreads();
  if (tid < C) {
    float s = 0.f;
    const float* gp = G + (size_t)b * N_ * C + tid;
    for (int k = 0; k < N_; ++k) s += arow[k] * gp[(size_t)k * C];
    out[((size_t)b * N_ + i) * C + tid] = s < 0.f ? 0.01f * s : s;
  }
}

__global__ void gh2_kernel(const float* Ho1, const float* Wh2, const float* bh2, float* Gh2) {
  int idx = blockIdx.x * 256 + threadIdx.x;  // < B*N*16
  int c = idx & 15, bn = idx >> 4;
  const float* hp = Ho1 + (size_t)bn * 32;
  float s = bh2[c];
  for (int k = 0; k < 32; ++k) s += hp[k] * Wh2[k * 16 + c];
  Gh2[idx] = s;
}

__global__ void __launch_bounds__(64) fc_kernel(const float* Ho, const float* W_fc,
                                                const float* b_fc, float* out) {
  int idx = blockIdx.x * 64 + threadIdx.x;   // < B*N*3
  int p = idx % 3;
  int bn = idx / 3;
  int b = bn >> 9, n = bn & 511;
  const float* hp = Ho + (size_t)bn * 16;
  float s = b_fc[p];
  for (int k = 0; k < 16; ++k) s += hp[k] * W_fc[k * 3 + p];
  out[((size_t)b * 3 + p) * N_ + n] = s;     // (B, PRED, N, 1)
}

// ---------------- launch ----------------

extern "C" void kernel_launch(void* const* d_in, const int* in_sizes, int n_in,
                              void* d_out, int out_size, void* d_ws, size_t ws_size,
                              hipStream_t stream) {
  (void)in_sizes; (void)n_in; (void)out_size; (void)ws_size;
  const float* x      = (const float*)d_in[0];
  const float* adjn   = (const float*)d_in[1];
  const float* mask   = (const float*)d_in[2];
  const float* Vs     = (const float*)d_in[3];
  const float* Ws     = (const float*)d_in[4];
  const float* Wp     = (const float*)d_in[5];
  const float* Wa     = (const float*)d_in[6];
  const float* bs     = (const float*)d_in[7];
  const float* W1     = (const float*)d_in[8];
  const float* b1     = (const float*)d_in[9];
  const float* W2     = (const float*)d_in[10];
  const float* b2     = (const float*)d_in[11];
  const float* W3     = (const float*)d_in[12];
  const float* b3     = (const float*)d_in[13];
  const float* W_ih   = (const float*)d_in[14];
  const float* W_hh   = (const float*)d_in[15];
  const float* b_ih   = (const float*)d_in[16];
  const float* b_hh   = (const float*)d_in[17];
  const float* W_attn = (const float*)d_in[18];
  const float* b_attn = (const float*)d_in[19];
  const float* Wh1    = (const float*)d_in[20];
  const float* bh1    = (const float*)d_in[21];
  const float* Wh2    = (const float*)d_in[22];
  const float* bh2    = (const float*)d_in[23];
  const float* W_fc   = (const float*)d_in[24];
  const float* b_fc   = (const float*)d_in[25];
  float* out = (float*)d_out;

  // workspace carving (~83 MB total), 256B-aligned
  char* w = (char*)d_ws;
  size_t off = 0;
  auto alloc = [&](size_t bytes) -> void* {
    void* p = w + off;
    off = (off + bytes + 255) & ~size_t(255);
    return p;
  };
  float*  dis  = (float*)alloc((size_t)B_ * T_ * N_ * 4);
  float*  u    = (float*)alloc(N_ * 4);
  float*  vbuf = (float*)alloc((size_t)B_ * 23 * N_ * 4);
  __bf16* Vsp  = (__bf16*)alloc((size_t)N_ * N_ * 2);
  __bf16* G1   = (__bf16*)alloc((size_t)B_ * T_ * N_ * 32 * 2);
  float*  H1   = (float*)alloc((size_t)B_ * T_ * N_ * 32 * 4);
  __bf16* G2   = (__bf16*)alloc((size_t)B_ * T_ * N_ * 32 * 2);
  float*  H2   = (float*)alloc((size_t)B_ * T_ * N_ * 32 * 4);
  __bf16* G3   = (__bf16*)alloc((size_t)B_ * T_ * N_ * 64 * 2);
  float*  H3   = (float*)alloc((size_t)B_ * T_ * N_ * 64 * 4);
  float*  Hs   = (float*)alloc((size_t)B_ * N_ * 64 * 4);
  float*  Ftb  = (float*)alloc((size_t)B_ * N_ * 64 * 4);
  float*  Fa   = (float*)alloc((size_t)B_ * N_ * 128 * 4);
  float*  Gh1  = (float*)alloc((size_t)B_ * N_ * 32 * 4);
  float*  Ho1  = (float*)alloc((size_t)B_ * N_ * 32 * 4);
  float*  Gh2  = (float*)alloc((size_t)B_ * N_ * 16 * 4);
  float*  Ho   = (float*)alloc((size_t)B_ * N_ * 16 * 4);

  // --- prep ---
  prep_u_kernel<<<1, 512, 0, stream>>>(Ws, Wp, Wa, u);
  pack_vs_kernel<<<(N_ * N_) / 256, 256, 0, stream>>>(Vs, Vsp);
  build_g1_kernel<<<(B_ * T_ * N_ * 32) / 256, 256, 0, stream>>>(x, W1, b1, G1);
  dis_kernel<<<B_ * T_, 256, 0, stream>>>(x, mask, dis);
  v_kernel<<<B_ * 23, 256, 0, stream>>>(x, mask, dis, u, vbuf);

  // --- H1: t=0 plain At, t=1..23 fused Pt (WMMA) ---
  gcn_prop_kernel<32><<<B_ * 1 * 32, 128, 0, stream>>>(x, mask, dis, G1, H1, 0, 1);
  pt_h1_kernel<<<B_ * 23 * 32, 256, 0, stream>>>(x, mask, dis, vbuf, bs, Vsp, G1, H1);

  // --- H2, H3 (WMMA propagation with on-the-fly At) ---
  build_g2_kernel<<<(B_ * T_ * N_) / 256, 256, 0, stream>>>(H1, W2, b2, G2);
  gcn_prop_kernel<32><<<B_ * T_ * 32, 128, 0, stream>>>(x, mask, dis, G2, H2, 0, T_);
  build_g3_kernel<<<(B_ * T_ * N_) / 256, 256, 0, stream>>>(H2, W3, b3, G3);
  gcn_prop_kernel<64><<<B_ * T_ * 32, 128, 0, stream>>>(x, mask, dis, G3, H3, 0, T_);
  mean_h3_kernel<<<(B_ * N_ * 64) / 256, 256, 0, stream>>>(H3, Hs);

  // --- LSTM branch ---
  lstm_kernel<<<B_ * N_, 64, 0, stream>>>(x, W_ih, W_hh, b_ih, b_hh, Ftb);

  // --- attention gate + hybrid GCN head ---
  attn_kernel<<<B_ * N_, 128, 0, stream>>>(Hs, Ftb, W_attn, b_attn, Fa);
  gh1_kernel<<<(B_ * N_ * 32) / 256, 256, 0, stream>>>(Fa, Wh1, bh1, Gh1);
  hprop_kernel<32><<<B_ * N_, 64, 0, stream>>>(adjn, Gh1, Ho1);
  gh2_kernel<<<(B_ * N_ * 16) / 256, 256, 0, stream>>>(Ho1, Wh2, bh2, Gh2);
  hprop_kernel<16><<<B_ * N_, 64, 0, stream>>>(adjn, Gh2, Ho);
  fc_kernel<<<(B_ * N_ * 3) / 64, 64, 0, stream>>>(Ho, W_fc, b_fc, out);
}
